// Head_61881888800788
// MI455X (gfx1250) — compile-verified, matrix operand order checked by reading, not run on hardware
//
#include <hip/hip_runtime.h>
#include <hip/hip_bf16.h>

// CDNA5 / gfx1250 attention head kernel. wave32, WMMA bf16 path.

typedef __attribute__((ext_vector_type(16))) __bf16 v16bf;
typedef __attribute__((ext_vector_type(8)))  float  v8f;
typedef __attribute__((ext_vector_type(4)))  float  v4f;

#define WMMA_BF16(A, B, C) \
  __builtin_amdgcn_wmma_f32_16x16x32_bf16(false, (A), false, (B), (short)0, (C), false, false)

__device__ __forceinline__ int kmap(int e, int half) {
  // 16-bit A/B fragment element -> K index (per ISA 7.12.2 layout tables)
  // lanes 0-15 : e 0..7 -> K 0..7,  e 8..15 -> K 16..23
  // lanes 16-31: e 0..7 -> K 8..15, e 8..15 -> K 24..31
  return e + ((e >> 3) << 3) + (half << 3);
}

constexpr int EMBD  = 1024;
constexpr int HEAD  = 64;
constexpr int SEQ   = 2048;
constexpr int BATCH = 8;
constexpr int QTILES = SEQ / 16;                       // 128
constexpr size_t ATTN_ELEMS = (size_t)BATCH * SEQ * SEQ;

// ---------------- f32 -> bf16 conversion ----------------
__global__ void cvt_f32_bf16(const float* __restrict__ src, __bf16* __restrict__ dst, int n) {
  int i = blockIdx.x * blockDim.x + threadIdx.x;
  int stride = gridDim.x * blockDim.x;
  for (; i < n; i += stride) dst[i] = (__bf16)src[i];
}

// ---------------- projection: y[M,64] = x[M,1024] @ W.T, * scale ----------------
// 256 threads = 8 waves; each wave owns a 16-row x tile and all 64 heads.
__global__ void proj_gemm(const __bf16* __restrict__ xb, const __bf16* __restrict__ wb,
                          __bf16* __restrict__ yb, float scale) {
  const int tid  = threadIdx.x;
  const int lane = tid & 31;
  const int wave = tid >> 5;
  const int half = lane >> 4;
  const int n    = lane & 15;
  const int rowbase = (blockIdx.x * 8 + wave) * 16;
  const int m = rowbase + n;

  v8f acc0 = {}, acc1 = {}, acc2 = {}, acc3 = {};

  for (int k0 = 0; k0 < EMBD; k0 += 32) {
    v16bf a;
#pragma unroll
    for (int e = 0; e < 16; ++e)
      a[e] = xb[(size_t)m * EMBD + k0 + kmap(e, half)];

    v16bf b0, b1, b2, b3;
#pragma unroll
    for (int e = 0; e < 16; ++e) {
      const int kk = k0 + kmap(e, half);
      b0[e] = wb[(size_t)(0 * 16 + n) * EMBD + kk];
      b1[e] = wb[(size_t)(1 * 16 + n) * EMBD + kk];
      b2[e] = wb[(size_t)(2 * 16 + n) * EMBD + kk];
      b3[e] = wb[(size_t)(3 * 16 + n) * EMBD + kk];
    }
    acc0 = WMMA_BF16(a, b0, acc0);
    acc1 = WMMA_BF16(a, b1, acc1);
    acc2 = WMMA_BF16(a, b2, acc2);
    acc3 = WMMA_BF16(a, b3, acc3);
  }

#pragma unroll
  for (int r = 0; r < 8; ++r) {
    const int row = rowbase + r + 8 * half;
    yb[(size_t)row * HEAD + 0 * 16 + n] = (__bf16)(acc0[r] * scale);
    yb[(size_t)row * HEAD + 1 * 16 + n] = (__bf16)(acc1[r] * scale);
    yb[(size_t)row * HEAD + 2 * 16 + n] = (__bf16)(acc2[r] * scale);
    yb[(size_t)row * HEAD + 3 * 16 + n] = (__bf16)(acc3[r] * scale);
  }
}

// ---------------- attention: one 16-query-row stripe per workgroup ----------------
// 128 threads = 4 waves. Full 16x2048 f32 score stripe lives in LDS (128 KB).
__global__ void attn_kernel(const __bf16* __restrict__ qbuf,
                            const __bf16* __restrict__ kbuf,
                            const __bf16* __restrict__ vbuf,
                            float* __restrict__ attn_out,
                            float* __restrict__ out) {
  __shared__ float s_p[16 * SEQ];          // 128 KB score / prob stripe
  __shared__ float s_redmax[8 * 16];
  __shared__ float s_redsum[8 * 16];
  __shared__ float s_rowmax[16];
  __shared__ float s_rinv[16];
  __shared__ float s_outbuf[4 * 16 * HEAD]; // 16 KB cross-wave reduce

  const int tid  = threadIdx.x;
  const int lane = tid & 31;
  const int wave = tid >> 5;
  const int half = lane >> 4;
  const int n    = lane & 15;

  const int b  = blockIdx.x >> 7;
  const int qt = blockIdx.x & 127;
  const int q0 = qt * 16;
  const int ncols = (qt + 1) * 16;

  const __bf16* Qb = qbuf + (size_t)b * SEQ * HEAD;
  const __bf16* Kb = kbuf + (size_t)b * SEQ * HEAD;
  const __bf16* Vb = vbuf + (size_t)b * SEQ * HEAD;

  // zero the stripe (masked / never-touched columns must read as p = 0)
  {
    v4f* p4 = (v4f*)s_p;
    const v4f z = {};
    for (int i = tid; i < (16 * SEQ) / 4; i += 128) p4[i] = z;
  }
  __syncthreads();

  // A fragments: 16x64 q tile (two 16x32 chunks)
  v16bf a0, a1;
  {
    const int mq = q0 + n;
#pragma unroll
    for (int e = 0; e < 16; ++e) {
      const int kk = kmap(e, half);
      a0[e] = Qb[(size_t)mq * HEAD + kk];
      a1[e] = Qb[(size_t)mq * HEAD + kk + 32];
    }
  }

  // ---- Phase A: scores s = q . k  (key tiles split across waves mod 4) ----
  for (int j = wave; j <= qt; j += 4) {
    // prefetch the next key tile this wave will touch (global_prefetch_b8)
    if (j + 4 <= qt)
      __builtin_prefetch(&Kb[(size_t)((j + 4) * 16 + n) * HEAD], 0, 3);

    v16bf b0, b1;
    const int key = j * 16 + n;
#pragma unroll
    for (int e = 0; e < 16; ++e) {
      const int kk = kmap(e, half);
      b0[e] = Kb[(size_t)key * HEAD + kk];
      b1[e] = Kb[(size_t)key * HEAD + kk + 32];
    }
    v8f c = {};
    c = WMMA_BF16(a0, b0, c);
    c = WMMA_BF16(a1, b1, c);

    if (j == qt) {  // causal mask on the diagonal tile only
#pragma unroll
      for (int r = 0; r < 8; ++r) {
        const int rowg = q0 + r + 8 * half;
        const int colg = j * 16 + n;
        if (colg > rowg) c[r] = -3.0e38f;
      }
    }
#pragma unroll
    for (int r = 0; r < 8; ++r)
      s_p[(r + 8 * half) * SEQ + j * 16 + n] = c[r];
  }
  __syncthreads();

  // ---- Phase B: row softmax over cols [0, ncols). 8 column-chunks x 16 rows ----
  const int srow  = tid & 15;
  const int chunk = tid >> 4;   // 0..7
  {
    float mx = -3.0e38f;
    for (int col = chunk; col < ncols; col += 8)
      mx = fmaxf(mx, s_p[srow * SEQ + col]);
    s_redmax[chunk * 16 + srow] = mx;
  }
  __syncthreads();
  if (tid < 16) {
    float mm = s_redmax[srow];
#pragma unroll
    for (int c = 1; c < 8; ++c) mm = fmaxf(mm, s_redmax[c * 16 + srow]);
    s_rowmax[srow] = mm;
  }
  __syncthreads();
  {
    const float rmax = s_rowmax[srow];
    float sum = 0.0f;
    for (int col = chunk; col < ncols; col += 8) {
      const float ev = __expf(s_p[srow * SEQ + col] - rmax);
      s_p[srow * SEQ + col] = ev;
      sum += ev;
    }
    s_redsum[chunk * 16 + srow] = sum;
  }
  __syncthreads();
  if (tid < 16) {
    float ss = 0.0f;
#pragma unroll
    for (int c = 0; c < 8; ++c) ss += s_redsum[c * 16 + srow];
    s_rinv[srow] = 1.0f / ss;
  }
  __syncthreads();

  // ---- Phase C1: fused normalize + stream to HBM ----
  // Write-once 134 MB stream: non-temporal b128 stores so it does not evict
  // the x/K/V working set from the 192 MB L2.
  {
    const v4f* lp4 = (const v4f*)s_p;
    v4f* sp4 = (v4f*)s_p;
    v4f* g4  = (v4f*)(attn_out + (size_t)(b * SEQ + q0) * SEQ);
    for (int i4 = tid; i4 < (16 * SEQ) / 4; i4 += 128) {
      const int row = i4 >> 9;                      // SEQ/4 == 512
      const float rv = s_rinv[row];
      v4f v = lp4[i4];
      v *= rv;
      sp4[i4] = v;                                  // normalized p back to LDS for P@V
      __builtin_nontemporal_store(v, &g4[(size_t)row * (SEQ / 4) + (i4 & 511)]);
    }
  }
  __syncthreads();

  // ---- Phase C2: out = P @ V, K split across waves in 32-key chunks ----
  v8f oc0 = {}, oc1 = {}, oc2 = {}, oc3 = {};
  const int nchunks = (ncols + 31) / 32;
  for (int ck = wave; ck < nchunks; ck += 4) {
    const int kb0 = ck * 32;
    v16bf ap;
#pragma unroll
    for (int e = 0; e < 16; ++e)
      ap[e] = (__bf16)s_p[n * SEQ + kb0 + kmap(e, half)];

    v16bf vb0, vb1, vb2, vb3;
#pragma unroll
    for (int e = 0; e < 16; ++e) {
      const int key = kb0 + kmap(e, half);
      vb0[e] = Vb[(size_t)key * HEAD + 0 * 16 + n];
      vb1[e] = Vb[(size_t)key * HEAD + 1 * 16 + n];
      vb2[e] = Vb[(size_t)key * HEAD + 2 * 16 + n];
      vb3[e] = Vb[(size_t)key * HEAD + 3 * 16 + n];
    }
    oc0 = WMMA_BF16(ap, vb0, oc0);
    oc1 = WMMA_BF16(ap, vb1, oc1);
    oc2 = WMMA_BF16(ap, vb2, oc2);
    oc3 = WMMA_BF16(ap, vb3, oc3);
  }
  __syncthreads();
#pragma unroll
  for (int r = 0; r < 8; ++r) {
    const int row = r + 8 * half;
    s_outbuf[wave * (16 * HEAD) + row * HEAD + 0 * 16 + n] = oc0[r];
    s_outbuf[wave * (16 * HEAD) + row * HEAD + 1 * 16 + n] = oc1[r];
    s_outbuf[wave * (16 * HEAD) + row * HEAD + 2 * 16 + n] = oc2[r];
    s_outbuf[wave * (16 * HEAD) + row * HEAD + 3 * 16 + n] = oc3[r];
  }
  __syncthreads();
  for (int i = tid; i < 16 * HEAD; i += 128) {
    const float s = s_outbuf[i] + s_outbuf[16 * HEAD + i] +
                    s_outbuf[2 * 16 * HEAD + i] + s_outbuf[3 * 16 * HEAD + i];
    const int row = i >> 6;
    const int col = i & 63;
    out[(size_t)(b * SEQ + q0 + row) * HEAD + col] = s;
  }
}

// ---------------- host ----------------
extern "C" void kernel_launch(void* const* d_in, const int* in_sizes, int n_in,
                              void* d_out, int out_size, void* d_ws, size_t ws_size,
                              hipStream_t stream) {
  (void)in_sizes; (void)n_in; (void)out_size; (void)ws_size;
  const float* x  = (const float*)d_in[0];
  const float* Wk = (const float*)d_in[1];
  const float* Wq = (const float*)d_in[2];
  const float* Wv = (const float*)d_in[3];
  float* outp = (float*)d_out;

  char* ws = (char*)d_ws;
  // workspace layout (bytes):
  //   xb : 8*2048*1024 bf16 = 33,554,432
  //   Wkb/Wqb/Wvb: 64*1024 bf16 = 131,072 each
  //   kb/qb/vb: 8*2048*64 bf16 = 2,097,152 each   (total ~40.2 MB)
  __bf16* xb  = (__bf16*)(ws);
  __bf16* wkb = (__bf16*)(ws + 33554432);
  __bf16* wqb = (__bf16*)(ws + 33554432 + 131072);
  __bf16* wvb = (__bf16*)(ws + 33554432 + 2 * 131072);
  __bf16* kb  = (__bf16*)(ws + 33554432 + 3 * 131072);
  __bf16* qb  = kb + (size_t)BATCH * SEQ * HEAD;
  __bf16* vb  = qb + (size_t)BATCH * SEQ * HEAD;

  cvt_f32_bf16<<<4096, 256, 0, stream>>>(x,  xb,  BATCH * SEQ * EMBD);
  cvt_f32_bf16<<<64,   256, 0, stream>>>(Wk, wkb, HEAD * EMBD);
  cvt_f32_bf16<<<64,   256, 0, stream>>>(Wq, wqb, HEAD * EMBD);
  cvt_f32_bf16<<<64,   256, 0, stream>>>(Wv, wvb, HEAD * EMBD);

  proj_gemm<<<128, 256, 0, stream>>>(xb, wkb, kb, 1.0f);
  proj_gemm<<<128, 256, 0, stream>>>(xb, wqb, qb, 1.0f);
  proj_gemm<<<128, 256, 0, stream>>>(xb, wvb, vb, 0.125f);   // HEAD_SIZE^-0.5 folded into v

  attn_kernel<<<BATCH * QTILES, 128, 0, stream>>>(qb, kb, vb, outp, outp + ATTN_ELEMS);
}
